// CrossGeoAtt_36292473651711
// MI455X (gfx1250) — matrix operand order repaired and checked by reading, not compiled, not run on hardware
//
#include <hip/hip_runtime.h>

// CDNA5 / gfx1250: wave32, WMMA 16x16x32 bf16 -> f32 accumulate.
typedef __attribute__((ext_vector_type(16))) __bf16 v16bf;
typedef __attribute__((ext_vector_type(8)))  float  v8f;
typedef __attribute__((ext_vector_type(4)))  int    v4i;

#if defined(__has_builtin)
#if __has_builtin(__builtin_amdgcn_global_load_async_to_lds_b128) && \
    __has_builtin(__builtin_amdgcn_s_wait_asynccnt)
#define USE_ASYNC_LDS 1
#endif
#endif
#ifndef USE_ASYNC_LDS
#define USE_ASYNC_LDS 0
#endif

#if USE_ASYNC_LDS
__device__ __forceinline__ void async_ld_b128(const void* g, void* l) {
  // AS1 (global) and AS3 (LDS) v4i* built via integer casts; the low 32
  // bits of a flat LDS pointer are the DS byte offset (ISA 10.2 aperture rule).
  __builtin_amdgcn_global_load_async_to_lds_b128(
      (__attribute__((address_space(1))) v4i*)(unsigned long long)g,
      (__attribute__((address_space(3))) v4i*)(unsigned)(unsigned long long)l,
      0, 0);
}
#endif

__device__ __forceinline__ unsigned short f2bf(float f) {
  union { float f; unsigned u; } v; v.f = f;
  unsigned r = v.u + 0x7FFFu + ((v.u >> 16) & 1u);  // round-to-nearest-even
  return (unsigned short)(r >> 16);
}

// ---------------- elementwise f32 -> bf16 ----------------
__global__ void k_cvt_bf16(const float* __restrict__ in, unsigned short* __restrict__ out, int n) {
  int i = blockIdx.x * 256 + threadIdx.x;
  if (i < n) out[i] = f2bf(in[i]);
}

// ------- tiled transpose: f32 (rows x cols) -> bf16 (cols x rows), batched over z -------
__global__ __launch_bounds__(256) void k_transpose_bf16(const float* __restrict__ in,
    unsigned short* __restrict__ out, int rows, int cols, long long inB, long long outB) {
  __shared__ float tile[32][33];
  in  += (size_t)blockIdx.z * inB;
  out += (size_t)blockIdx.z * outB;
  int r0 = blockIdx.y * 32, c0 = blockIdx.x * 32;
  int tx = threadIdx.x & 31, ty = threadIdx.x >> 5;  // ty: 0..7
#pragma unroll
  for (int i = 0; i < 32; i += 8)
    tile[ty + i][tx] = in[(size_t)(r0 + ty + i) * cols + (c0 + tx)];
  __syncthreads();
#pragma unroll
  for (int i = 0; i < 32; i += 8)
    out[(size_t)(c0 + ty + i) * rows + (r0 + tx)] = f2bf(tile[tx][ty + i]);
}

// ---------------- NT GEMM: C[m,n] = sum_k A[m,k]*B[n,k] (+bias[m]) ----------------
// A: M x K bf16 row-major.  B: N x K bf16 row-major.  Block tile 128x128, BK=32.
// 8 wave32 waves; wave tile 32(M) x 64(N) = 2x4 WMMA 16x16 tiles.
// LDS fill: gfx1250 async global->LDS (double-buffered, 1 barrier/iter) when the
// builtin exists; otherwise global->VGPR->ds_store (2 barriers/iter).
// Fragment loads follow ISA 7.12.2:
//   A 16x32 bf16: lane L holds row L%16; elems 0..7 = K (L/16)*8..+7, elems 8..15 = +16.
//   B 32x16 bf16: lane L holds col L%16; elems 0..7 = K (L/16)*8..+7, elems 8..15 = +16 (same as A).
template<bool OUT_BF16, bool HAS_BIAS>
__global__ __launch_bounds__(256) void k_gemm_nt(
    const unsigned short* __restrict__ A, const unsigned short* __restrict__ B,
    void* __restrict__ Cout, const float* __restrict__ bias,
    int M, int N, int K, long long aB, long long bB, long long cB)
{
  (void)M;
  constexpr int LDSW = 40;  // row stride (elems): 80B, 16B-aligned, bank-spread
#if USE_ASYNC_LDS
  __shared__ unsigned short sA[2][128 * LDSW];
  __shared__ unsigned short sB[2][128 * LDSW];
#else
  __shared__ unsigned short sA[1][128 * LDSW];
  __shared__ unsigned short sB[1][128 * LDSW];
#endif
  A += (size_t)blockIdx.z * aB;
  B += (size_t)blockIdx.z * bB;

  const int tid = threadIdx.x, lane = tid & 31, wid = tid >> 5;
  const int wm = (wid >> 1) * 32, wn = (wid & 1) * 64;
  const int m0 = blockIdx.y * 128, n0 = blockIdx.x * 128;
  const int lrow = lane & 15, lhi = lane >> 4;

  const int fr = tid >> 1;          // LDS fill: tile row 0..127
  const int fc = (tid & 1) * 16;    // 16 bf16 per thread (two 16B chunks)
  const unsigned short* gA = A + (size_t)(m0 + fr) * K + fc;
  const unsigned short* gB = B + (size_t)(n0 + fr) * K + fc;

  v8f acc[2][4] = {};

#if USE_ASYNC_LDS
  unsigned short* lA[2] = { &sA[0][fr * LDSW + fc], &sA[1][fr * LDSW + fc] };
  unsigned short* lB[2] = { &sB[0][fr * LDSW + fc], &sB[1][fr * LDSW + fc] };

  // prologue: tile 0 into buffer 0
  async_ld_b128(gA, lA[0]); async_ld_b128(gA + 8, lA[0] + 8);
  async_ld_b128(gB, lB[0]); async_ld_b128(gB + 8, lB[0] + 8);
  gA += 32; gB += 32;

  const int nIter = K >> 5;
  int cur = 0;
  for (int it = 0; it < nIter; ++it) {
    __builtin_amdgcn_s_wait_asynccnt(0);  // my writes to buf[cur] landed
    __syncthreads();                      // all writes visible; prior reads retired
    if (it + 1 < nIter) {                 // prefetch next tile, overlaps compute
      int nx = cur ^ 1;
      async_ld_b128(gA, lA[nx]); async_ld_b128(gA + 8, lA[nx] + 8);
      async_ld_b128(gB, lB[nx]); async_ld_b128(gB + 8, lB[nx] + 8);
      gA += 32; gB += 32;
    }
    const unsigned short* bA = sA[cur];
    const unsigned short* bBp = sB[cur];
#else
  unsigned short* lA = &sA[0][fr * LDSW + fc];
  unsigned short* lB = &sB[0][fr * LDSW + fc];
  for (int k0 = 0; k0 < K; k0 += 32) {
    uint4 a0 = *(const uint4*)(gA);
    uint4 a1 = *(const uint4*)(gA + 8);
    uint4 b0 = *(const uint4*)(gB);
    uint4 b1 = *(const uint4*)(gB + 8);
    gA += 32; gB += 32;
    __syncthreads();
    *(uint4*)(lA) = a0; *(uint4*)(lA + 8) = a1;
    *(uint4*)(lB) = b0; *(uint4*)(lB + 8) = b1;
    __syncthreads();
    const unsigned short* bA = sA[0];
    const unsigned short* bBp = sB[0];
#endif

    union Frag { uint4 u[2]; v16bf v; } af[2], bfr[4];
#pragma unroll
    for (int mi = 0; mi < 2; ++mi) {
      const unsigned short* q = &bA[(wm + mi * 16 + lrow) * LDSW + lhi * 8];
      af[mi].u[0] = *(const uint4*)(q);
      af[mi].u[1] = *(const uint4*)(q + 16);
    }
#pragma unroll
    for (int ni = 0; ni < 4; ++ni) {
      const unsigned short* q = &bBp[(wn + ni * 16 + lrow) * LDSW + lhi * 8];
      bfr[ni].u[0] = *(const uint4*)(q);
      bfr[ni].u[1] = *(const uint4*)(q + 16);
    }
#pragma unroll
    for (int mi = 0; mi < 2; ++mi)
#pragma unroll
      for (int ni = 0; ni < 4; ++ni)
        acc[mi][ni] = __builtin_amdgcn_wmma_f32_16x16x32_bf16(
            false, af[mi].v, false, bfr[ni].v, (short)0, acc[mi][ni], false, false);

#if USE_ASYNC_LDS
    cur ^= 1;
  }
#else
  }
#endif

  // C layout (ISA 7.12.2): VGPR v -> row v + 8*(lane/16); col = lane%16.
#pragma unroll
  for (int mi = 0; mi < 2; ++mi)
#pragma unroll
    for (int ni = 0; ni < 4; ++ni)
#pragma unroll
      for (int v = 0; v < 8; ++v) {
        int row = m0 + wm + mi * 16 + lhi * 8 + v;
        int col = n0 + wn + ni * 16 + lrow;
        float val = acc[mi][ni][v];
        if (HAS_BIAS) val += bias[row];
        size_t off = (size_t)blockIdx.z * cB + (size_t)row * N + col;
        if (OUT_BF16) ((unsigned short*)Cout)[off] = f2bf(val);
        else          ((float*)Cout)[off]          = val;
      }
}

// ------- 2x2 maxpool of (1024, 64, 64) f32 -> bf16 in both (c,q) and (q,c) layouts -------
__global__ void k_maxpool_dual(const float* __restrict__ C1,
                               unsigned short* __restrict__ F,
                               unsigned short* __restrict__ FT) {
  size_t z = blockIdx.z;
  int i = blockIdx.x * 256 + threadIdx.x;     // 0 .. 1024*1024-1
  int c = i >> 10, q = i & 1023;
  int qy = q >> 5, qx = q & 31;
  const float* p = C1 + z * (size_t)(1024 * 4096) + (size_t)c * 4096 + (qy * 2) * 64 + qx * 2;
  float m = fmaxf(fmaxf(p[0], p[1]), fmaxf(p[64], p[65]));
  unsigned short bm = f2bf(m);
  F [z * 1048576u + (size_t)c * 1024 + q] = bm;
  FT[z * 1048576u + (size_t)q * 1024 + c] = bm;
}

// ------- relu + row L2-normalize (axis = last, 1024 wide), f32 in -> bf16 out -------
__global__ __launch_bounds__(256) void k_relu_rownorm(const float* __restrict__ fm,
                                                      unsigned short* __restrict__ Mout) {
  size_t base = (size_t)blockIdx.z * 1048576u + (size_t)blockIdx.x * 1024u;
  const float* row = fm + base;
  int tid = threadIdx.x, lane = tid & 31, wid = tid >> 5;
  float v[4], s = 0.f;
#pragma unroll
  for (int j = 0; j < 4; ++j) {
    float x = fmaxf(row[tid + j * 256], 0.f);
    v[j] = x; s += x * x;
  }
#pragma unroll
  for (int o = 16; o; o >>= 1) s += __shfl_xor(s, o, 32);
  __shared__ float red[8];
  if (lane == 0) red[wid] = s;
  __syncthreads();
  s = red[0] + red[1] + red[2] + red[3] + red[4] + red[5] + red[6] + red[7];
  float inv = rsqrtf(s + 1e-6f);
#pragma unroll
  for (int j = 0; j < 4; ++j) Mout[base + tid + j * 256] = f2bf(v[j] * inv);
}

// ------- 2x bilinear upsample, align_corners: (256,32,32) -> (256,64,64), f32 -------
__global__ void k_upsample2(const float* __restrict__ in, float* __restrict__ out) {
  size_t z = blockIdx.z;
  int i = blockIdx.x * 256 + threadIdx.x;     // 0 .. 256*4096-1
  int c = i >> 12, r = i & 4095;
  int oy = r >> 6, ox = r & 63;
  const float sc = 31.0f / 63.0f;
  float fy = oy * sc, fx = ox * sc;
  int y0 = (int)fy, x0 = (int)fx;
  float ty = fy - y0, tx = fx - x0;
  int y1 = min(y0 + 1, 31), x1 = min(x0 + 1, 31);
  const float* p = in + z * (size_t)(256 * 1024) + (size_t)c * 1024;
  float v00 = p[y0 * 32 + x0], v01 = p[y0 * 32 + x1];
  float v10 = p[y1 * 32 + x0], v11 = p[y1 * 32 + x1];
  out[z * (size_t)(256 * 4096) + i] =
      (1.f - ty) * ((1.f - tx) * v00 + tx * v01) + ty * ((1.f - tx) * v10 + tx * v11);
}

extern "C" void kernel_launch(void* const* d_in, const int* in_sizes, int n_in,
                              void* d_out, int out_size, void* d_ws, size_t ws_size,
                              hipStream_t stream) {
  (void)in_sizes; (void)n_in; (void)out_size; (void)ws_size;
  const float* xa  = (const float*)d_in[0];
  const float* xb  = (const float*)d_in[1];
  const float* w1  = (const float*)d_in[2];
  const float* b1  = (const float*)d_in[3];
  const float* We  = (const float*)d_in[4];
  const float* We2 = (const float*)d_in[5];
  const float* w2  = (const float*)d_in[6];
  const float* b2  = (const float*)d_in[7];
  float* out = (float*)d_out;

  char* wsp = (char*)d_ws;
  auto take = [&](size_t bytes) -> void* {
    void* r = (void*)wsp;
    wsp += (bytes + 255) & ~(size_t)255;
    return r;
  };
  const size_t NB = 8;
  const long long M1 = 1024LL * 1024LL;       // 1M-element batch stride
  unsigned short* Wb   = (unsigned short*)take(1024 * 256 * 2);
  unsigned short* Web  = (unsigned short*)take(1024 * 1024 * 2);
  unsigned short* We2b = (unsigned short*)take(1024 * 1024 * 2);
  unsigned short* W2b  = (unsigned short*)take(256 * 1024 * 2);
  unsigned short* XaT  = (unsigned short*)take(NB * 4096 * 256 * 2);
  unsigned short* XbT  = (unsigned short*)take(NB * 4096 * 256 * 2);
  unsigned short* Fa   = (unsigned short*)take(NB * M1 * 2);
  unsigned short* FaT  = (unsigned short*)take(NB * M1 * 2);
  unsigned short* Fb   = (unsigned short*)take(NB * M1 * 2);
  unsigned short* FbT  = (unsigned short*)take(NB * M1 * 2);
  unsigned short* Pm   = (unsigned short*)take(NB * M1 * 2);
  unsigned short* Mab  = (unsigned short*)take(NB * M1 * 2);
  unsigned short* Mba  = (unsigned short*)take(NB * M1 * 2);
  unsigned short* Ha   = (unsigned short*)take(NB * 256 * 1024 * 2);
  unsigned short* Hb   = (unsigned short*)take(NB * 256 * 1024 * 2);
  float* FM  = (float*)take(NB * M1 * 4);
  float* O32 = (float*)take(NB * 256 * 1024 * 4);
  float* C1  = (float*)take(NB * 1024 * 4096 * 4);   // 128 MB conv scratch

  dim3 blk(256);
  // weights -> bf16
  k_cvt_bf16<<<dim3(1024), blk, 0, stream>>>(w1, Wb, 1024 * 256);
  k_cvt_bf16<<<dim3(4096), blk, 0, stream>>>(We, Web, 1024 * 1024);
  k_cvt_bf16<<<dim3(4096), blk, 0, stream>>>(We2, We2b, 1024 * 1024);
  k_cvt_bf16<<<dim3(1024), blk, 0, stream>>>(w2, W2b, 256 * 1024);
  // X^T (sp,c) bf16 for conv GEMM B-operand
  k_transpose_bf16<<<dim3(128, 8, NB), blk, 0, stream>>>(xa, XaT, 256, 4096, 256LL * 4096, 4096LL * 256);
  k_transpose_bf16<<<dim3(128, 8, NB), blk, 0, stream>>>(xb, XbT, 256, 4096, 256LL * 4096, 4096LL * 256);

  // conv1 (+bias) then 2x2 maxpool -> Fa/FaT, Fb/FbT   (C1 reused for side b)
  k_gemm_nt<false, true><<<dim3(32, 8, NB), blk, 0, stream>>>(Wb, XaT, C1, b1,
      1024, 4096, 256, 0, 4096LL * 256, 1024LL * 4096);
  k_maxpool_dual<<<dim3(4096, 1, NB), blk, 0, stream>>>(C1, Fa, FaT);
  k_gemm_nt<false, true><<<dim3(32, 8, NB), blk, 0, stream>>>(Wb, XbT, C1, b1,
      1024, 4096, 256, 0, 4096LL * 256, 1024LL * 4096);
  k_maxpool_dual<<<dim3(4096, 1, NB), blk, 0, stream>>>(C1, Fb, FbT);

  // Mab: P = FbT * We^T ; fm = P * FaT^T ; relu+rownorm
  k_gemm_nt<true, false><<<dim3(8, 8, NB), blk, 0, stream>>>(FbT, Web, Pm, nullptr,
      1024, 1024, 1024, M1, 0, M1);
  k_gemm_nt<false, false><<<dim3(8, 8, NB), blk, 0, stream>>>(Pm, FaT, FM, nullptr,
      1024, 1024, 1024, M1, M1, M1);
  k_relu_rownorm<<<dim3(1024, 1, NB), blk, 0, stream>>>(FM, Mab);
  // Mba
  k_gemm_nt<true, false><<<dim3(8, 8, NB), blk, 0, stream>>>(FaT, We2b, Pm, nullptr,
      1024, 1024, 1024, M1, 0, M1);
  k_gemm_nt<false, false><<<dim3(8, 8, NB), blk, 0, stream>>>(Pm, FbT, FM, nullptr,
      1024, 1024, 1024, M1, M1, M1);
  k_relu_rownorm<<<dim3(1024, 1, NB), blk, 0, stream>>>(FM, Mba);

  // side a: H = conv2_w * Mab^T ; out32 = H * Fa^T + b2 ; upsample (conv/upsample commuted)
  k_gemm_nt<true, false><<<dim3(8, 2, NB), blk, 0, stream>>>(W2b, Mab, Ha, nullptr,
      256, 1024, 1024, 0, M1, 256LL * 1024);
  k_gemm_nt<false, true><<<dim3(8, 2, NB), blk, 0, stream>>>(Ha, Fa, O32, b2,
      256, 1024, 1024, 256LL * 1024, M1, 256LL * 1024);
  k_upsample2<<<dim3(4096, 1, NB), blk, 0, stream>>>(O32, out);
  // side b
  k_gemm_nt<true, false><<<dim3(8, 2, NB), blk, 0, stream>>>(W2b, Mba, Hb, nullptr,
      256, 1024, 1024, 0, M1, 256LL * 1024);
  k_gemm_nt<false, true><<<dim3(8, 2, NB), blk, 0, stream>>>(Hb, Fb, O32, b2,
      256, 1024, 1024, 256LL * 1024, M1, 256LL * 1024);
  k_upsample2<<<dim3(4096, 1, NB), blk, 0, stream>>>(O32, out + (size_t)NB * 256 * 4096);
}
